// TriAttStart_20194936225860
// MI455X (gfx1250) — matrix operand order repaired and checked by reading, not compiled, not run on hardware
//
#include <hip/hip_runtime.h>
#include <hip/hip_bf16.h>
#include <math.h>

// ---------------- CDNA5 WMMA types/helpers ----------------
typedef __bf16 bf16x16 __attribute__((ext_vector_type(16)));
typedef float  f32x8   __attribute__((ext_vector_type(8)));

static constexpr int LL   = 512;  // sequence length (both axes)
static constexpr int ZD   = 128;  // z channel dim
static constexpr int NH   = 4;    // heads
static constexpr int CC   = 32;   // head dim
static constexpr int HC   = 128;  // NH*CC
static constexpr int WPAD = 136;  // LDS row stride (halfs): 272B, 16B aligned, bank-rotating

__device__ __forceinline__ f32x8 wmma_bf16(bf16x16 a, bf16x16 b, f32x8 c) {
  return __builtin_amdgcn_wmma_f32_16x16x32_bf16(false, a, false, b, (short)0, c,
                                                 false, false);
}

// A-matrix fragment (16x32, MxK), row pointer = &A[m][0], K-contiguous, 16B-aligned.
// Per ISA layout: half 0 -> K 0..7 & 16..23, half 1 -> K 8..15 & 24..31.
__device__ __forceinline__ bf16x16 load_fragA(const __bf16* row, int hf) {
  union { bf16x16 v; uint4 q[2]; } f;
  const char* p = (const char*)row + hf * 16;
  f.q[0] = *(const uint4*)(p);
  f.q[1] = *(const uint4*)(p + 32);
  return f.v;
}
// B-matrix fragment (32x16, KxN) from transposed storage Bt[n][k], 16B-aligned rows.
// Same K order as A: half 0 -> K 0..7 & 16..23, half 1 -> K 8..15 & 24..31.
__device__ __forceinline__ bf16x16 load_fragB(const __bf16* row, int hf) {
  union { bf16x16 v; uint4 q[2]; } f;
  const char* p = (const char*)row + hf * 16;
  f.q[0] = *(const uint4*)(p);
  f.q[1] = *(const uint4*)(p + 32);
  return f.v;
}

// ---------------- CDNA5 async global->LDS copy (ASYNCcnt path) ----------------
// lds_off: byte offset into workgroup LDS (dynamic LDS starts at 0 here).
__device__ __forceinline__ void async_b128(unsigned int lds_off, const void* g) {
  asm volatile("global_load_async_to_lds_b128 %0, %1, off"
               :: "v"(lds_off), "v"((unsigned long long)(size_t)g)
               : "memory");
}
template <int N>
__device__ __forceinline__ void wait_async() {
#if __has_builtin(__builtin_amdgcn_s_wait_asynccnt)
  __builtin_amdgcn_s_wait_asynccnt((short)N);
#else
  asm volatile("s_wait_asynccnt %0" :: "i"(N) : "memory");
#endif
}

// ---------------- Kernel 1: LayerNorm + q/k/v/g/bias projections ----------------
// grid = 4096 (one block per (i1, 64-wide i2 tile)), block = 256 (8 waves)
__global__ __launch_bounds__(256) void k_proj(
    const float* __restrict__ z, const float* __restrict__ lng,
    const float* __restrict__ lnb,
    const float* __restrict__ qw, const float* __restrict__ qb,
    const float* __restrict__ kw, const float* __restrict__ kb,
    const float* __restrict__ vw, const float* __restrict__ vb,
    const float* __restrict__ bw, const float* __restrict__ bb,
    const float* __restrict__ gw, const float* __restrict__ gb,
    __bf16* __restrict__ q_ws, __bf16* __restrict__ k_ws,
    __bf16* __restrict__ vT, __bf16* __restrict__ g_ws,
    float* __restrict__ bias_ws) {
  extern __shared__ char smem[];
  __bf16* zt    = (__bf16*)smem;                 // 64 * WPAD halfs
  __bf16* wT    = zt + 64 * WPAD;                // 4 * 128 * WPAD halfs
  float*  red   = (float*)(wT + 4 * 128 * WPAD); // 512 floats
  float*  stats = red + 512;                     // 128 floats (mu, rstd)

  const int tid = threadIdx.x;
  const int bid = blockIdx.x;
  const int i1 = bid >> 3;
  const int j0 = (bid & 7) << 6;

  // Transpose all 4 projection weights into LDS bf16: wT[p][n][k] = w[k][n].
#pragma unroll
  for (int p = 0; p < 4; ++p) {
    const float* w = (p == 0) ? qw : (p == 1) ? kw : (p == 2) ? vw : gw;
    for (int it = 0; it < 64; ++it) {
      int e = tid + (it << 8);      // 0..16383
      int k = e >> 7, n = e & 127;
      wT[(p * 128 + n) * WPAD + k] = (__bf16)w[(k << 7) + n];
    }
  }

  // LayerNorm partial sums: 4 threads per row.
  {
    int jj = tid >> 2, s = tid & 3;
    const float* zr = z + ((size_t)i1 * LL + j0 + jj) * ZD + s * 32;
    float sm = 0.f, sq = 0.f;
#pragma unroll 8
    for (int k = 0; k < 32; ++k) { float x = zr[k]; sm += x; sq += x * x; }
    red[jj * 4 + s] = sm;
    red[256 + jj * 4 + s] = sq;
  }
  __syncthreads();
  if (tid < 64) {
    float sm = red[tid * 4] + red[tid * 4 + 1] + red[tid * 4 + 2] + red[tid * 4 + 3];
    float sq = red[256 + tid * 4] + red[256 + tid * 4 + 1] +
               red[256 + tid * 4 + 2] + red[256 + tid * 4 + 3];
    float mu  = sm * (1.f / 128.f);
    float var = sq * (1.f / 128.f) - mu * mu;
    stats[tid]      = mu;
    stats[64 + tid] = rsqrtf(var + 1e-5f);
  }
  __syncthreads();
  // Normalized z tile -> LDS bf16.
  for (int it = 0; it < 32; ++it) {
    int e = tid + (it << 8);        // 0..8191
    int jj = e >> 7, k = e & 127;
    float x = z[((size_t)i1 * LL + j0 + jj) * ZD + k];
    zt[jj * WPAD + k] = (__bf16)((x - stats[jj]) * stats[64 + jj] * lng[k] + lnb[k]);
  }
  __syncthreads();

  // Pair bias projection (128 -> 4), one dot per thread.
  {
    int jj = tid >> 2, h = tid & 3;
    float acc = bb[h];
    const __bf16* zr = zt + jj * WPAD;
#pragma unroll 8
    for (int k = 0; k < 128; ++k) acc += (float)zr[k] * bw[(k << 2) + h];
    bias_ws[((size_t)i1 * LL + j0 + jj) * NH + h] = acc;
  }

  // WMMA GEMMs: [64 x 128] x [128 x 128] for p in {q,k,v,g}.
  const int lane = tid & 31, wv = tid >> 5;
  const int lo = lane & 15, hf = lane >> 4;
  for (int it = 0; it < 16; ++it) {
    int st = wv + (it << 3);                 // 0..127
    int p = st >> 5, mt = (st >> 3) & 3, nt = st & 7;
    f32x8 acc = {};
    const __bf16* arow = zt + (mt * 16 + lo) * WPAD;
    const __bf16* brow = wT + (p * 128 + nt * 16 + lo) * WPAD;
#pragma unroll
    for (int kc = 0; kc < 4; ++kc)
      acc = wmma_bf16(load_fragA(arow + kc * 32, hf),
                      load_fragB(brow + kc * 32, hf), acc);
    int col = nt * 16 + lo;
    const float* pb = (p == 0) ? qb : (p == 1) ? kb : (p == 2) ? vb : gb;
    float cb = pb[col];
    int mbase = mt * 16 + hf * 8;
    if (p == 2) {  // v: store transposed vT[h][b][c][j]
      int h = col >> 5, c = col & 31;
      __bf16* dst = vT + (((size_t)(h * LL + i1)) * CC + c) * LL + j0 + mbase;
#pragma unroll
      for (int r = 0; r < 8; ++r) dst[r] = (__bf16)(acc[r] + cb);
    } else {
      __bf16* base = (p == 0) ? q_ws : (p == 1) ? k_ws : g_ws;
      __bf16* dst = base + ((size_t)(i1 * LL + j0 + mbase)) * HC + col;
#pragma unroll
      for (int r = 0; r < 8; ++r) dst[(size_t)r * HC] = (__bf16)(acc[r] + cb);
    }
  }
}

// ---------------- Kernel 2: att[h][l][k] = scale * sum_b q.k + bias ----------------
// grid = 4*8*8 = 256 blocks (64x64 tile per head), block = 256 (8 waves).
// Double-buffered async staging: per stage, 2 b-rows (K=64) of q (64x64 halfs)
// and k (64x64 halfs) land in LDS; WMMA reads fragments from LDS.
static constexpr int ATT_RS   = 72;              // LDS tile row stride in halfs (144B)
static constexpr int ATT_BUF  = 64 * ATT_RS * 2; // 9216 B per tile buffer
__global__ __launch_bounds__(256) void k_att(
    const __bf16* __restrict__ q_ws, const __bf16* __restrict__ k_ws,
    const float* __restrict__ bias_ws, float* __restrict__ att) {
  extern __shared__ char smem[];
  // layout: qbuf0 @0, qbuf1 @9216, kbuf0 @18432, kbuf1 @27648 (36,864 B total)
  const int bid = blockIdx.x;
  const int h = bid >> 6, t = bid & 63;
  const int l0 = (t >> 3) << 6, k0 = (t & 7) << 6;
  const int tid = threadIdx.x, lane = tid & 31, wv = tid >> 5;
  const int lo = lane & 15, hf = lane >> 4;
  const int mt = wv & 3, nt0 = (wv >> 2) << 1;

  // Staging assignment: thread -> (row rc, 16B segment seg); i in {0,1} = b-row.
  const int seg = tid & 3, rc = tid >> 2;
  const size_t STEP = (size_t)2 * LL * HC;  // halfs per stage (2 b-rows)
  const __bf16* gq0 = q_ws + ((size_t)(l0 + rc)) * HC + h * CC + seg * 8;
  const __bf16* gq1 = gq0 + (size_t)LL * HC;
  const __bf16* gk0 = k_ws + ((size_t)(k0 + rc)) * HC + h * CC + seg * 8;
  const __bf16* gk1 = gk0 + (size_t)LL * HC;
  const unsigned laq = (unsigned)(rc * 144 + seg * 16);
  const unsigned lak = (unsigned)(2 * ATT_BUF + rc * 144 + seg * 16);

  auto issue4 = [&](int par) {
    unsigned pb = (unsigned)par * ATT_BUF;
    async_b128(laq + pb,      gq0);
    async_b128(laq + pb + 64, gq1);
    async_b128(lak + pb,      gk0);
    async_b128(lak + pb + 64, gk1);
    gq0 += STEP; gq1 += STEP; gk0 += STEP; gk1 += STEP;
  };

  f32x8 acc0 = {}, acc1 = {};
  issue4(0);
  for (int s = 0; s < 256; ++s) {
    if (s < 255) { issue4((s + 1) & 1); wait_async<4>(); }
    else         { wait_async<0>(); }
    __syncthreads();
    const __bf16* qb_ = (const __bf16*)smem + (s & 1) * (ATT_BUF / 2);
    const __bf16* kb_ = (const __bf16*)(smem + 2 * ATT_BUF) + (s & 1) * (ATT_BUF / 2);
    const __bf16* ar  = qb_ + (mt * 16 + lo) * ATT_RS;
    const __bf16* br0 = kb_ + (nt0 * 16 + lo) * ATT_RS;
    const __bf16* br1 = br0 + 16 * ATT_RS;
#pragma unroll
    for (int c2 = 0; c2 < 2; ++c2) {
      bf16x16 a = load_fragA(ar + c2 * 32, hf);
      acc0 = wmma_bf16(a, load_fragB(br0 + c2 * 32, hf), acc0);
      acc1 = wmma_bf16(a, load_fragB(br1 + c2 * 32, hf), acc1);
    }
    __syncthreads();
  }

  const float scale = 1.0f / 128.0f;  // (1/sqrt(32))/sqrt(512)
#pragma unroll
  for (int s = 0; s < 2; ++s) {
    f32x8 acc = s ? acc1 : acc0;
    int kidx = k0 + (nt0 + s) * 16 + lo;
    int lbase = l0 + mt * 16 + hf * 8;
#pragma unroll
    for (int r = 0; r < 8; ++r) {
      int l = lbase + r;
      att[((size_t)h * LL + l) * LL + kidx] =
          acc[r] * scale + bias_ws[((size_t)l * LL + kidx) * NH + h];
    }
  }
}

// ---------------- Kernel 3: softmax over k (one wave per (h,l) row) ----------------
__global__ __launch_bounds__(256) void k_softmax(const float* __restrict__ att,
                                                 __bf16* __restrict__ attP) {
  const int row  = blockIdx.x * 8 + (threadIdx.x >> 5);
  const int lane = threadIdx.x & 31;
  const float* src = att + (size_t)row * LL;
  float x[16], mx = -3.0e38f;
#pragma unroll
  for (int i = 0; i < 16; ++i) { x[i] = src[lane + 32 * i]; mx = fmaxf(mx, x[i]); }
#pragma unroll
  for (int d = 16; d >= 1; d >>= 1) mx = fmaxf(mx, __shfl_xor(mx, d, 32));
  float sum = 0.f;
#pragma unroll
  for (int i = 0; i < 16; ++i) { x[i] = __expf(x[i] - mx); sum += x[i]; }
#pragma unroll
  for (int d = 16; d >= 1; d >>= 1) sum += __shfl_xor(sum, d, 32);
  const float inv = 1.0f / sum;
  __bf16* dst = attP + (size_t)row * LL;
#pragma unroll
  for (int i = 0; i < 16; ++i) dst[lane + 32 * i] = (__bf16)(x[i] * inv);
}

// ---------------- Kernel 4: o = att @ v, gated by sigmoid(g) ----------------
// grid = 512*4*8 = 16384, block = 256 (8 waves, one 16x16 subtile each).
// v tile (32 c x 512 k) staged to LDS once via async copies.
static constexpr int AV_RS = 520;  // halfs (1040B rows, 16B aligned)
__global__ __launch_bounds__(256) void k_av(
    const __bf16* __restrict__ attP, const __bf16* __restrict__ vT,
    const __bf16* __restrict__ g_ws, __bf16* __restrict__ og) {
  extern __shared__ char smem[];  // 32 * 1040 = 33,280 B
  const int bid = blockIdx.x;
  const int b = bid >> 5, rest = bid & 31;
  const int h = rest >> 3, l0 = (rest & 7) << 6;
  const int tid = threadIdx.x, lane = tid & 31, wv = tid >> 5;
  const int lo = lane & 15, hf = lane >> 4;
  const int mt = wv & 3, nt = wv >> 2;

  // Stage v[b,h]: 32 rows x 1024B from vT[h][b][c][k] into LDS.
  const __bf16* vbase = vT + ((size_t)(h * LL + b)) * CC * LL;
#pragma unroll
  for (int i = 0; i < 8; ++i) {
    int e = tid + (i << 8);          // 0..2047
    int c = e >> 6, sg = e & 63;
    async_b128((unsigned)(c * 1040 + sg * 16), vbase + (size_t)c * LL + sg * 8);
  }
  wait_async<0>();
  __syncthreads();

  f32x8 acc = {};
  const __bf16* arow = attP + ((size_t)h * LL + l0 + mt * 16 + lo) * LL;
  const __bf16* brow = (const __bf16*)smem + (nt * 16 + lo) * AV_RS;
#pragma unroll 4
  for (int kc = 0; kc < 16; ++kc)
    acc = wmma_bf16(load_fragA(arow + kc * 32, hf),
                    load_fragB(brow + kc * 32, hf), acc);
  const int c = nt * 16 + lo;
  const int lbase = l0 + mt * 16 + hf * 8;
#pragma unroll
  for (int r = 0; r < 8; ++r) {
    int l = lbase + r;
    size_t idx = ((size_t)(b * LL + l)) * HC + h * CC + c;
    float gv = (float)g_ws[idx];
    float sg = 1.0f / (1.0f + __expf(-gv));
    og[idx] = (__bf16)(acc[r] * sg);
  }
}

// ---------------- Kernel 5: output projection [262144x128]@[128x128] + ob ----------------
__global__ __launch_bounds__(256) void k_oproj(
    const __bf16* __restrict__ og, const float* __restrict__ ow,
    const float* __restrict__ ob, float* __restrict__ out) {
  extern __shared__ char smem[];
  __bf16* owT = (__bf16*)smem;  // 128 * WPAD halfs
  const int tid = threadIdx.x;
  for (int it = 0; it < 64; ++it) {
    int e = tid + (it << 8);
    int k = e >> 7, n = e & 127;
    owT[n * WPAD + k] = (__bf16)ow[(k << 7) + n];
  }
  __syncthreads();
  const size_t p0 = (size_t)blockIdx.x * 64;
  const int lane = tid & 31, wv = tid >> 5;
  const int lo = lane & 15, hf = lane >> 4;
  for (int it = 0; it < 4; ++it) {
    int st = wv + (it << 3);
    int mt = st >> 3, nt = st & 7;
    f32x8 acc = {};
    const __bf16* arow = og + (p0 + mt * 16 + lo) * HC;
    const __bf16* brow = owT + (nt * 16 + lo) * WPAD;
#pragma unroll
    for (int kc = 0; kc < 4; ++kc)
      acc = wmma_bf16(load_fragA(arow + kc * 32, hf),
                      load_fragB(brow + kc * 32, hf), acc);
    int col = nt * 16 + lo;
    float cb = ob[col];
    int mbase = mt * 16 + hf * 8;
    float* dst = out + (p0 + mbase) * HC + col;
#pragma unroll
    for (int r = 0; r < 8; ++r) dst[(size_t)r * HC] = acc[r] + cb;
  }
}

// ---------------- Host launch ----------------
extern "C" void kernel_launch(void* const* d_in, const int* in_sizes, int n_in,
                              void* d_out, int out_size, void* d_ws, size_t ws_size,
                              hipStream_t stream) {
  (void)in_sizes; (void)n_in; (void)out_size; (void)ws_size;
  const float* z   = (const float*)d_in[0];
  const float* lng = (const float*)d_in[1];
  const float* lnb = (const float*)d_in[2];
  const float* qw  = (const float*)d_in[3];
  const float* qb  = (const float*)d_in[4];
  const float* kw  = (const float*)d_in[5];
  const float* kb  = (const float*)d_in[6];
  const float* vw  = (const float*)d_in[7];
  const float* vb  = (const float*)d_in[8];
  const float* bw  = (const float*)d_in[9];
  const float* bb  = (const float*)d_in[10];
  const float* gw  = (const float*)d_in[11];
  const float* gb  = (const float*)d_in[12];
  const float* ow  = (const float*)d_in[13];
  const float* ob  = (const float*)d_in[14];

  char* ws = (char*)d_ws;
  auto take = [&ws](size_t bytes) {
    char* p = ws;
    ws += (bytes + 255) & ~(size_t)255;
    return p;
  };
  const size_t nPos = (size_t)LL * LL;
  float*  bias_ws = (float*)take(nPos * NH * sizeof(float));
  __bf16* q_ws    = (__bf16*)take(nPos * HC * sizeof(__bf16));
  __bf16* k_ws    = (__bf16*)take(nPos * HC * sizeof(__bf16));
  __bf16* vT      = (__bf16*)take(nPos * HC * sizeof(__bf16));
  __bf16* g_ws    = (__bf16*)take(nPos * HC * sizeof(__bf16));
  float*  att     = (float*)take((size_t)NH * LL * LL * sizeof(float));
  __bf16* attP    = (__bf16*)take((size_t)NH * LL * LL * sizeof(__bf16));
  __bf16* og      = q_ws;  // alias: q is dead after k_att

  const size_t smem1 = (64 * WPAD + 4 * 128 * WPAD) * sizeof(__bf16) +
                       (512 + 128) * sizeof(float);          // 159,232 B
  const size_t smem2 = 4 * (size_t)ATT_BUF;                  // 36,864 B
  const size_t smem4 = 32 * (size_t)AV_RS * sizeof(__bf16);  // 33,280 B
  const size_t smem5 = 128 * (size_t)WPAD * sizeof(__bf16);  // 34,816 B

  k_proj<<<4096, 256, smem1, stream>>>(z, lng, lnb, qw, qb, kw, kb, vw, vb, bw,
                                       bb, gw, gb, q_ws, k_ws, vT, g_ws, bias_ws);
  k_att<<<256, 256, smem2, stream>>>(q_ws, k_ws, bias_ws, att);
  k_softmax<<<256, 256, 0, stream>>>(att, attP);
  k_av<<<16384, 256, smem4, stream>>>(attP, vT, g_ws, og);
  k_oproj<<<4096, 256, smem5, stream>>>(og, ow, ob, (float*)d_out);
}
